// DGL_GIN_52175262712464
// MI455X (gfx1250) — matrix operand order repaired and checked by reading, not compiled, not run on hardware
//
#include <hip/hip_runtime.h>
#include <hip/hip_bf16.h>

typedef __attribute__((ext_vector_type(16))) __bf16 v16bf;
typedef __attribute__((ext_vector_type(8)))  __bf16 v8bf;
typedef __attribute__((ext_vector_type(4)))  __bf16 v4bf;
typedef __attribute__((ext_vector_type(8)))  float  v8f;
typedef __attribute__((ext_vector_type(4)))  int    v4i;
typedef __attribute__((address_space(1)))    v4i    gv4i;   // global int4
typedef __attribute__((address_space(3)))    v4i    lv4i;   // LDS int4

#define N_NODES 100000
#define N_EDGES 1600000
#define G_GRAPHS 128
#define D_IN 20
#define D_EMB 300
#define D_HID 600
#define D_OUT 64
#define SD  320   // padded stride for 300 (K multiple of 32)
#define SH  640   // padded stride for 600
#define SD0 32    // padded stride for layer-0 input (20)
#define BN_EPS 1e-5f

#if defined(__has_builtin)
#if __has_builtin(__builtin_amdgcn_global_load_async_to_lds_b128) && \
    __has_builtin(__builtin_amdgcn_s_wait_asynccnt)
#define HAVE_ASYNC_LDS 1
#endif
#endif

// ---------------- utility kernels ----------------

__global__ void zero_f32_kernel(float* __restrict__ p, long n) {
  long i = (long)blockIdx.x * blockDim.x + threadIdx.x;
  if (i < n) p[i] = 0.f;
}

// dst[n*KP + k] = src[k*Nn + n] (transpose to column-contiguous), zero-padded, bf16
__global__ void pack_w_kernel(const float* __restrict__ src, __bf16* __restrict__ dst,
                              int K, int Nn, int KP, int NP) {
  int idx = blockIdx.x * blockDim.x + threadIdx.x;
  if (idx >= KP * NP) return;
  int n = idx / KP, k = idx - n * KP;
  float v = (k < K && n < Nn) ? src[k * Nn + n] : 0.f;
  dst[idx] = (__bf16)v;
}

__global__ void copy_x0_kernel(const float* __restrict__ nf, float* __restrict__ x) {
  int idx = blockIdx.x * blockDim.x + threadIdx.x;
  if (idx >= N_NODES * SD0) return;
  int r = idx >> 5, c = idx & 31;
  x[idx] = (c < D_IN) ? nf[r * D_IN + c] : 0.f;
}

// ---------------- edge scatter: agg[dst] += x[src] + e ----------------

__global__ void scatter_kernel(const float* __restrict__ x, const float* __restrict__ ef,
                               const int* __restrict__ src, const int* __restrict__ dst,
                               float* __restrict__ agg, int d, int sd) {
  int e = blockIdx.x;
  int f = threadIdx.x;
  if (f >= d) return;
  int s = src[e], t = dst[e];
  float v = x[(size_t)s * sd + f] + ef[e];
  atomicAdd(&agg[(size_t)t * sd + f], v);
}

// ---------------- WMMA GEMM: C = act(A @ W + bias) ----------------
// A: [nRows, KP] row-major (f32 or bf16), contiguous (ld == KP).
// Bt: [NP, KP] bf16 (i.e. W^T, column-of-W contiguous).
// Block: 256 threads = 8 waves; tile = 16 rows x 128 cols; wave = 16x16 tile.
// A-tile is staged into LDS as bf16 (converted once during staging for the f32 case).

template<bool AF32, bool RELU_BF16_OUT>
__global__ __launch_bounds__(256)
void gemm_kernel(const void* __restrict__ Av, const __bf16* __restrict__ Bt,
                 const float* __restrict__ bias, void* __restrict__ Cv,
                 int KP, int NP, int Nn, int ldC) {
  __shared__ __align__(16) unsigned char smem[16 * SH * 2];  // 16 rows x KP bf16 (<= 20480 B)
  const int tid = threadIdx.x;
  const int rowbase = blockIdx.y * 16;

  if (AF32) {
    // Stage + convert: fp32 tile -> bf16 tile in LDS (one conversion, shared by all waves).
    const float4* s = (const float4*)((const float*)Av + (size_t)rowbase * KP);
    v4bf* dv = (v4bf*)smem;
    int n = 16 * KP / 4;
    for (int i = tid; i < n; i += 256) {
      float4 f = s[i];
      v4bf o;
      o[0] = (__bf16)f.x; o[1] = (__bf16)f.y; o[2] = (__bf16)f.z; o[3] = (__bf16)f.w;
      dv[i] = o;
    }
  } else {
    // Pure byte move: use the CDNA5 async-to-LDS path (ASYNCcnt) when available.
#if defined(HAVE_ASYNC_LDS)
    char* gsrc = (char*)Av + (size_t)rowbase * KP * 2;
    int n = 16 * KP * 2 / 16;  // 16-byte chunks
    for (int i = tid; i < n; i += 256) {
      __builtin_amdgcn_global_load_async_to_lds_b128(
          (gv4i*)(gsrc + (size_t)i * 16),
          (lv4i*)(smem + i * 16),
          /*offset=*/0, /*cpol=*/0);
    }
    __builtin_amdgcn_s_wait_asynccnt(0);
#else
    const uint4* s = (const uint4*)((const __bf16*)Av + (size_t)rowbase * KP);
    uint4* dv = (uint4*)smem;
    int n = 16 * KP / 8;
    for (int i = tid; i < n; i += 256) dv[i] = s[i];
#endif
  }
  __syncthreads();

  const int wave = tid >> 5;
  const int lane = tid & 31;
  const int colbase = blockIdx.x * 128 + wave * 16;
  if (colbase >= NP) return;              // whole-wave uniform exit: EXEC all-ones for WMMA

  const int lr = lane & 15;               // A row within tile / B-C column within tile
  const int hi = lane >> 4;
  const __bf16* sA = (const __bf16*)smem;
  const __bf16* arow = sA + lr * KP + hi * 16;
  const __bf16* bcol = Bt + (size_t)(colbase + lr) * KP + hi * 16;

  v8f acc = {};
  for (int kb = 0; kb < KP; kb += 32) {
    // A fragment: element e holds K = kb + 16*hi + e  (matches B)
    v8bf a0 = *(const v8bf*)(arow + kb);
    v8bf a1 = *(const v8bf*)(arow + kb + 8);
    v16bf a = __builtin_shufflevector(a0, a1, 0, 1, 2, 3, 4, 5, 6, 7,
                                      8, 9, 10, 11, 12, 13, 14, 15);
    // B fragment: lane holds col N=lr, K = kb + 16*hi + e  -> 16 contiguous bf16 (32B)
    v16bf b = *(const v16bf*)(bcol + kb);
    acc = __builtin_amdgcn_wmma_f32_16x16x32_bf16(false, a, false, b, (short)0, acc,
                                                  false, false);
  }

  const int gcol = colbase + lr;
  if (gcol >= Nn) return;
  float bv = bias[gcol];
#pragma unroll
  for (int r = 0; r < 8; ++r) {           // C VGPR r -> row = r + 8*hi (ISA C layout)
    int grow = rowbase + r + hi * 8;
    float v = acc[r] + bv;
    if (RELU_BF16_OUT) {
      v = v > 0.f ? v : 0.f;
      ((__bf16*)Cv)[(size_t)grow * ldC + gcol] = (__bf16)v;
    } else {
      ((float*)Cv)[(size_t)grow * ldC + gcol] = v;
    }
  }
}

// ---------------- batch-norm ----------------

__global__ void bn_reduce_kernel(const float* __restrict__ h, float* __restrict__ csum,
                                 float* __restrict__ csq, int nrows, int rpb) {
  int col = threadIdx.x;
  if (col >= D_EMB) return;
  int r0 = blockIdx.x * rpb;
  int r1 = min(r0 + rpb, nrows);
  float s = 0.f, q = 0.f;
  for (int r = r0; r < r1; ++r) {
    float v = h[(size_t)r * SD + col];
    s += v; q += v * v;
  }
  atomicAdd(&csum[col], s);
  atomicAdd(&csq[col], q);
}

__global__ void bn_finalize_kernel(const float* __restrict__ csum, const float* __restrict__ csq,
                                   const float* __restrict__ gamma, const float* __restrict__ beta,
                                   float* __restrict__ aS, float* __restrict__ bS) {
  int c = threadIdx.x;
  if (c >= D_EMB) return;
  float invn = 1.f / (float)N_NODES;
  float mu  = csum[c] * invn;
  float var = csq[c] * invn - mu * mu;
  float inv = rsqrtf(var + BN_EPS);
  float a = gamma[c] * inv;
  aS[c] = a;
  bS[c] = beta[c] - mu * a;
}

__global__ void bn_apply_kernel(const float* __restrict__ h, const float* __restrict__ aS,
                                const float* __restrict__ bS, float* __restrict__ x) {
  long idx = (long)blockIdx.x * blockDim.x + threadIdx.x;
  if (idx >= (long)N_NODES * SD) return;
  int col = (int)(idx % SD);
  x[idx] = (col < D_EMB) ? h[idx] * aS[col] + bS[col] : 0.f;  // keep pads zeroed
}

// ---------------- readout ----------------

__global__ void count_kernel(const int* __restrict__ gid, int* __restrict__ cnt) {
  int i = blockIdx.x * blockDim.x + threadIdx.x;
  if (i < N_NODES) atomicAdd(&cnt[gid[i]], 1);
}

__global__ void pool_kernel(const float* __restrict__ x, const int* __restrict__ gid,
                            float* __restrict__ gsum, int rpb) {
  int col = threadIdx.x;
  if (col >= D_EMB) return;
  int r0 = blockIdx.x * rpb;
  int r1 = min(r0 + rpb, N_NODES);
  if (r0 >= r1) return;
  float acc = 0.f;
  int cur = gid[r0];
  for (int r = r0; r < r1; ++r) {         // graph_ids are sorted: accumulate runs
    int g = gid[r];
    if (g != cur) { atomicAdd(&gsum[(size_t)cur * SD + col], acc); acc = 0.f; cur = g; }
    acc += x[(size_t)r * SD + col];
  }
  atomicAdd(&gsum[(size_t)cur * SD + col], acc);
}

__global__ void final_kernel(const float* __restrict__ gsum, const int* __restrict__ cnt,
                             const float* __restrict__ wt, const float* __restrict__ bt,
                             float* __restrict__ out) {
  int g = blockIdx.x, o = threadIdx.x;
  float inv = 1.f / fmaxf((float)cnt[g], 1.f);
  float acc = bt[o];
  for (int k = 0; k < D_EMB; ++k)
    acc += gsum[(size_t)g * SD + k] * inv * wt[k * D_OUT + o];
  out[g * D_OUT + o] = acc;
}

// ---------------- orchestration ----------------

extern "C" void kernel_launch(void* const* d_in, const int* in_sizes, int n_in,
                              void* d_out, int out_size, void* d_ws, size_t ws_size,
                              hipStream_t stream) {
  const float* nf      = (const float*)d_in[0];
  const float* ef      = (const float*)d_in[1];
  const int*   src     = (const int*)d_in[2];
  const int*   dst     = (const int*)d_in[3];
  const int*   gid     = (const int*)d_in[4];
  const float* w1_0    = (const float*)d_in[5];
  const float* b1_0    = (const float*)d_in[6];
  const float* w2_0    = (const float*)d_in[7];
  const float* b2_0    = (const float*)d_in[8];
  const float* gamma_0 = (const float*)d_in[9];
  const float* beta_0  = (const float*)d_in[10];
  const float* w1s     = (const float*)d_in[11];
  const float* b1s     = (const float*)d_in[12];
  const float* w2s     = (const float*)d_in[13];
  const float* b2s     = (const float*)d_in[14];
  const float* gammas  = (const float*)d_in[15];
  const float* betas   = (const float*)d_in[16];
  const float* wt      = (const float*)d_in[17];
  const float* bt      = (const float*)d_in[18];
  float* out = (float*)d_out;

  char* ws = (char*)d_ws;
  size_t off = 0;
  auto alloc = [&](size_t bytes) -> void* {
    void* p = ws + off;
    off += (bytes + 255) & ~(size_t)255;
    return p;
  };
  float*  xbuf = (float*)alloc((size_t)N_NODES * SD * 4);   // node features / BN output
  float*  hb   = (float*)alloc((size_t)N_NODES * SD * 4);   // agg, then MLP2 output
  __bf16* h1   = (__bf16*)alloc((size_t)N_NODES * SH * 2);  // relu(MLP1) in bf16
  __bf16* w1t0 = (__bf16*)alloc((size_t)SH * SD0 * 2);
  __bf16* w1t[4]; for (int l = 0; l < 4; ++l) w1t[l] = (__bf16*)alloc((size_t)SH * SD * 2);
  __bf16* w2t[5]; for (int l = 0; l < 5; ++l) w2t[l] = (__bf16*)alloc((size_t)SD * SH * 2);
  float* csum = (float*)alloc(SH * 4);   // csum & csq contiguous (zeroed together)
  float* csq  = (float*)alloc(SH * 4);
  float* aS   = (float*)alloc(SH * 4);
  float* bS   = (float*)alloc(SH * 4);
  float* gsum = (float*)alloc((size_t)G_GRAPHS * SD * 4);   // gsum & gcnt contiguous
  int*   gcnt = (int*)alloc(G_GRAPHS * 4);

  // Pack weights (transposed, padded, bf16).
  {
    int tot = SH * SD0;
    pack_w_kernel<<<(tot + 255) / 256, 256, 0, stream>>>(w1_0, w1t0, D_IN, D_HID, SD0, SH);
  }
  for (int l = 0; l < 4; ++l) {
    int tot = SH * SD;
    pack_w_kernel<<<(tot + 255) / 256, 256, 0, stream>>>(
        w1s + (size_t)l * D_EMB * D_HID, w1t[l], D_EMB, D_HID, SD, SH);
  }
  {
    int tot = SD * SH;
    pack_w_kernel<<<(tot + 255) / 256, 256, 0, stream>>>(w2_0, w2t[0], D_HID, D_EMB, SH, SD);
    for (int l = 1; l < 5; ++l)
      pack_w_kernel<<<(tot + 255) / 256, 256, 0, stream>>>(
          w2s + (size_t)(l - 1) * D_HID * D_EMB, w2t[l], D_HID, D_EMB, SH, SD);
  }

  // Layer-0 input, padded 20 -> 32.
  {
    int tot = N_NODES * SD0;
    copy_x0_kernel<<<(tot + 255) / 256, 256, 0, stream>>>(nf, xbuf);
  }

  for (int l = 0; l < 5; ++l) {
    const int d  = (l == 0) ? D_IN : D_EMB;
    const int sd = (l == 0) ? SD0 : SD;
    const __bf16* W1 = (l == 0) ? w1t0 : w1t[l - 1];
    const __bf16* W2 = w2t[l];
    const float* B1 = (l == 0) ? b1_0    : b1s    + (size_t)(l - 1) * D_HID;
    const float* B2 = (l == 0) ? b2_0    : b2s    + (size_t)(l - 1) * D_EMB;
    const float* GM = (l == 0) ? gamma_0 : gammas + (size_t)(l - 1) * D_EMB;
    const float* BT = (l == 0) ? beta_0  : betas  + (size_t)(l - 1) * D_EMB;

    long nagg = (long)N_NODES * sd;
    zero_f32_kernel<<<(int)((nagg + 255) / 256), 256, 0, stream>>>(hb, nagg);
    scatter_kernel<<<N_EDGES, (l == 0) ? 32 : 320, 0, stream>>>(xbuf, ef, src, dst, hb, d, sd);

    dim3 g1(SH / 128, N_NODES / 16);
    gemm_kernel<true, true><<<g1, 256, 0, stream>>>(hb, W1, B1, h1, sd, SH, D_HID, SH);

    zero_f32_kernel<<<(2 * SH + 255) / 256, 256, 0, stream>>>(csum, 2 * SH);

    dim3 g2((SD + 127) / 128, N_NODES / 16);
    gemm_kernel<false, false><<<g2, 256, 0, stream>>>(h1, W2, B2, hb, SH, SD, D_EMB, SD);

    const int rpb = 512;
    bn_reduce_kernel<<<(N_NODES + rpb - 1) / rpb, 320, 0, stream>>>(hb, csum, csq, N_NODES, rpb);
    bn_finalize_kernel<<<1, 320, 0, stream>>>(csum, csq, GM, BT, aS, bS);
    long nx = (long)N_NODES * SD;
    bn_apply_kernel<<<(int)((nx + 255) / 256), 256, 0, stream>>>(hb, aS, bS, xbuf);
  }

  // Readout.
  {
    long nz = (long)G_GRAPHS * SD + G_GRAPHS;  // gsum + gcnt (contiguous)
    zero_f32_kernel<<<(int)((nz + 255) / 256), 256, 0, stream>>>(gsum, nz);
  }
  count_kernel<<<(N_NODES + 255) / 256, 256, 0, stream>>>(gid, gcnt);
  pool_kernel<<<(N_NODES + 511) / 512, 320, 0, stream>>>(xbuf, gid, gsum, 512);
  final_kernel<<<G_GRAPHS, 64, 0, stream>>>(gsum, gcnt, wt, bt, out);
}
